// SCAN_43439299232414
// MI455X (gfx1250) — matrix operand order repaired and checked by reading, not compiled, not run on hardware
//
#include <hip/hip_runtime.h>
#include <stdint.h>

typedef __attribute__((ext_vector_type(16))) __bf16        v16bf;
typedef __attribute__((ext_vector_type(8)))  float         v8f;
typedef __attribute__((ext_vector_type(8)))  unsigned int  v8u;

#define B_ROWS   256
#define DIM      256
#define NBANK    100000
#define TILE     16
#define NTILES   (NBANK / TILE)   // 6250
#define NBLK     250
#define TPB      (NTILES / NBLK)  // 25 tiles per block
#define BPAD     264              // padded LDS row stride (bf16 elems) -> bank-conflict free

#define ALLONES64 0xFFFFFFFFFFFFFFFFull

__device__ __forceinline__ float bf_hi_f(float x) {
    return __uint_as_float(__float_as_uint(x) & 0xFFFF0000u);
}
__device__ __forceinline__ unsigned int pack_hi2(float x0, float x1) {
    return (__float_as_uint(x0) >> 16) | ((__float_as_uint(x1) >> 16) << 16);
}
__device__ __forceinline__ unsigned int pack_lo2(float x0, float x1) {
    float r0 = x0 - bf_hi_f(x0);
    float r1 = x1 - bf_hi_f(x1);
    return (__float_as_uint(r0) >> 16) | ((__float_as_uint(r1) >> 16) << 16);
}
__device__ __forceinline__ unsigned long long shflx16_u64(unsigned long long v, int mask) {
    unsigned int lo = (unsigned int)v;
    unsigned int hi = (unsigned int)(v >> 32);
    lo = __shfl_xor(lo, mask, 16);   // width 16: reduce within each 16-lane group
    hi = __shfl_xor(hi, mask, 16);
    return ((unsigned long long)hi << 32) | (unsigned long long)lo;
}

// ---------------------------------------------------------------------------
// Kernel 0: init per-row best keys to +inf (all ones)
// ---------------------------------------------------------------------------
__global__ void SCAN_init_kernel(unsigned long long* ws) {
    ws[threadIdx.x] = ALLONES64;  // 512 entries: [0..255]=primary, [256..511]=fallback
}

// ---------------------------------------------------------------------------
// Kernel 1: tiled bf16 hi/lo WMMA GEMM + masked running argmin, software-
// pipelined bank-tile loads (regs stage tile t+1 while tile t computes).
//   grid = NBLK x 512 threads (16 waves; wave w owns feature rows 16w..16w+15,
//   and cooperatively loads bank row w of each tile)
// ---------------------------------------------------------------------------
__global__ void __launch_bounds__(512, 1)
SCAN_main_kernel(const float* __restrict__ feature,
                 const float* __restrict__ bank,
                 const int*   __restrict__ cluster_label_bank,
                 const int*   __restrict__ class_label_bank,
                 const int*   __restrict__ cluster_idx,
                 const int*   __restrict__ gt_label,
                 unsigned long long* __restrict__ wsP,
                 unsigned long long* __restrict__ wsF) {
    __shared__ __align__(16) unsigned short Bhi[TILE][BPAD];
    __shared__ __align__(16) unsigned short Blo[TILE][BPAD];
    __shared__ float a2s[B_ROWS];
    __shared__ float b2s[TILE];
    __shared__ int   cidx_s[B_ROWS];
    __shared__ int   gts_s[B_ROWS];
    __shared__ int   clb_s[TILE];
    __shared__ int   csb_s[TILE];

    const int tid  = threadIdx.x;
    const int wave = tid >> 5;    // wave w loads bank row w; owns feature rows 16w..
    const int lane = tid & 31;
    const int half = lane >> 4;   // 0: lanes 0-15, 1: lanes 16-31
    const int nl   = lane & 15;

    // ---- stage 0: per-row |a|^2 and per-row labels into LDS (once) ----
    if (tid < B_ROWS) {
        const float4* fr = (const float4*)(feature + (size_t)tid * DIM);
        float s = 0.f;
        #pragma unroll 8
        for (int i = 0; i < DIM / 4; ++i) {
            float4 v = fr[i];
            s += v.x * v.x + v.y * v.y + v.z * v.z + v.w * v.w;
        }
        a2s[tid]    = s;
        cidx_s[tid] = cluster_idx[tid];
        gts_s[tid]  = gt_label[tid];
    }

    // ---- A fragments in registers (16-bit A 16x32, contiguous-K per lane half, hi/lo split) ----
    // lanes 0-15: M=lane,    K elems = {32c+0..15}
    // lanes 16-31: M=lane-16, K elems = {32c+16..31}
    v16bf Ah[8], Al[8];
    {
        const int    mrow = wave * TILE + nl;
        const float* arow = feature + (size_t)mrow * DIM;
        #pragma unroll
        for (int c = 0; c < 8; ++c) {
            const int k0 = 32 * c + 16 * half;       // first 8 contiguous K
            const int k1 = k0 + 8;                   // second 8 contiguous K
            float g[16];
            float4 q;
            q = *(const float4*)(arow + k0);     g[0]=q.x; g[1]=q.y; g[2]=q.z; g[3]=q.w;
            q = *(const float4*)(arow + k0 + 4); g[4]=q.x; g[5]=q.y; g[6]=q.z; g[7]=q.w;
            q = *(const float4*)(arow + k1);     g[8]=q.x; g[9]=q.y; g[10]=q.z; g[11]=q.w;
            q = *(const float4*)(arow + k1 + 4); g[12]=q.x; g[13]=q.y; g[14]=q.z; g[15]=q.w;
            v8u hu, lu;
            #pragma unroll
            for (int v = 0; v < 8; ++v) {
                hu[v] = pack_hi2(g[2 * v], g[2 * v + 1]);
                lu[v] = pack_lo2(g[2 * v], g[2 * v + 1]);
            }
            Ah[c] = __builtin_bit_cast(v16bf, hu);
            Al[c] = __builtin_bit_cast(v16bf, lu);
        }
    }

    // running (dist_bits << 32 | idx) minima, 8 rows per lane (C-matrix layout rows)
    unsigned long long bestP[8], bestF[8];
    #pragma unroll
    for (int r = 0; r < 8; ++r) { bestP[r] = ALLONES64; bestF[r] = ALLONES64; }

    const int t0 = blockIdx.x * TPB;
    const int t1 = t0 + TPB;

    // ---- pipeline prologue: stage tile t0 into registers ----
    float4 pq0, pq1;
    int    pclb = 0, pcsb = 0;
    {
        const int    n0  = t0 * TILE;
        const float* src = bank + (size_t)(n0 + wave) * DIM + lane * 8;
        pq0 = *(const float4*)src;
        pq1 = *(const float4*)(src + 4);
        if (lane == 0) {
            pclb = cluster_label_bank[n0 + wave];
            pcsb = class_label_bank[n0 + wave];
        }
    }

    for (int t = t0; t < t1; ++t) {
        const int n0 = t * TILE;

        __syncthreads();   // previous compute done reading LDS tile / b2s / labels

        // ---- store stage: convert staged regs -> bf16 hi/lo LDS; wave-reduce |b|^2 ----
        {
            float x[8] = {pq0.x, pq0.y, pq0.z, pq0.w, pq1.x, pq1.y, pq1.z, pq1.w};
            unsigned int hp[4], lp[4];
            float ssq = 0.f;
            #pragma unroll
            for (int j = 0; j < 4; ++j) {
                hp[j] = pack_hi2(x[2 * j], x[2 * j + 1]);
                lp[j] = pack_lo2(x[2 * j], x[2 * j + 1]);
                ssq += x[2 * j] * x[2 * j] + x[2 * j + 1] * x[2 * j + 1];
            }
            const int col = lane * 8;
            *(uint4*)&Bhi[wave][col] = make_uint4(hp[0], hp[1], hp[2], hp[3]);
            *(uint4*)&Blo[wave][col] = make_uint4(lp[0], lp[1], lp[2], lp[3]);
            #pragma unroll
            for (int off = 16; off >= 1; off >>= 1)
                ssq += __shfl_xor(ssq, off, 32);     // full-wave sum (row == wave)
            if (lane == 0) {
                b2s[wave]  = ssq;
                clb_s[wave] = pclb;
                csb_s[wave] = pcsb;
            }
        }

        // ---- prefetch stage: issue global loads for tile t+1 (overlap WMMA) ----
        {
            const int    tn  = (t + 1 < t1) ? t + 1 : t;   // clamp; last iter redundant
            const int    nn0 = tn * TILE;
            const float* src = bank + (size_t)(nn0 + wave) * DIM + lane * 8;
            pq0 = *(const float4*)src;
            pq1 = *(const float4*)(src + 4);
            if (lane == 0) {
                pclb = cluster_label_bank[nn0 + wave];
                pcsb = class_label_bank[nn0 + wave];
            }
        }

        __syncthreads();   // LDS tile t ready

        // ---- 16x16 tile GEMM: 8 K-chunks x 3 bf16 WMMAs (hh + lh + hl) ----
        v8f acc = {};
        #pragma unroll
        for (int c = 0; c < 8; ++c) {
            // B fragment (32x16): lane = N col, halves split K 0-15 / 16-31,
            // VGPR v holds K = 2v, 2v+1 -> 16 contiguous bf16 in LDS row
            const int koff = 32 * c + 16 * half;
            const uint4* ph = (const uint4*)&Bhi[nl][koff];
            const uint4* pl = (const uint4*)&Blo[nl][koff];
            v8u bhu, blu;
            ((uint4*)&bhu)[0] = ph[0]; ((uint4*)&bhu)[1] = ph[1];
            ((uint4*)&blu)[0] = pl[0]; ((uint4*)&blu)[1] = pl[1];
            v16bf bh = __builtin_bit_cast(v16bf, bhu);
            v16bf bl = __builtin_bit_cast(v16bf, blu);
            acc = __builtin_amdgcn_wmma_f32_16x16x32_bf16(false, Ah[c], false, bh,
                                                          (short)0, acc, false, false);
            acc = __builtin_amdgcn_wmma_f32_16x16x32_bf16(false, Al[c], false, bh,
                                                          (short)0, acc, false, false);
            acc = __builtin_amdgcn_wmma_f32_16x16x32_bf16(false, Ah[c], false, bl,
                                                          (short)0, acc, false, false);
        }

        // ---- epilogue: dist + masks + running min (C layout: VGPR r -> M=r+8*half) ----
        const float b2v = b2s[nl];
        const int   cl  = clb_s[nl];
        const int   cs  = csb_s[nl];
        const unsigned long long key_lo = (unsigned long long)(unsigned int)(n0 + nl);
        #pragma unroll
        for (int r = 0; r < 8; ++r) {
            const int m = wave * TILE + r + 8 * half;
            float d2   = a2s[m] + b2v - 2.0f * acc[r];
            float dist = sqrtf(fmaxf(d2, 0.0f));
            unsigned long long key =
                ((unsigned long long)__float_as_uint(dist) << 32) | key_lo;
            bool dc = (cs != gts_s[m]);
            bool sc = (cl == cidx_s[m]);
            unsigned long long kf = dc ? key : ALLONES64;
            unsigned long long kp = (dc && sc) ? key : ALLONES64;
            bestF[r] = (kf < bestF[r]) ? kf : bestF[r];
            bestP[r] = (kp < bestP[r]) ? kp : bestP[r];
        }
    }

    // ---- reduce across the 16 lanes sharing each row, then global atomic min ----
    #pragma unroll
    for (int r = 0; r < 8; ++r) {
        unsigned long long p = bestP[r];
        unsigned long long f = bestF[r];
        #pragma unroll
        for (int off = 8; off >= 1; off >>= 1) {
            unsigned long long pp = shflx16_u64(p, off);
            unsigned long long ff = shflx16_u64(f, off);
            p = (pp < p) ? pp : p;
            f = (ff < f) ? ff : f;
        }
        if (nl == 0) {
            const int m = wave * TILE + r + 8 * half;
            atomicMin(&wsP[m], p);
            atomicMin(&wsF[m], f);
        }
    }
}

// ---------------------------------------------------------------------------
// Kernel 2: pick primary-if-any else fallback, gather bank row
// ---------------------------------------------------------------------------
__global__ void SCAN_gather_kernel(const float* __restrict__ bank,
                                   const unsigned long long* __restrict__ wsP,
                                   const unsigned long long* __restrict__ wsF,
                                   float* __restrict__ out) {
    const int b = blockIdx.x;
    const unsigned long long p = wsP[b];
    const unsigned long long f = wsF[b];
    unsigned long long best = (p != ALLONES64) ? p : f;
    unsigned int idx = (best == ALLONES64) ? 0u : (unsigned int)best;  // argmin-of-all-inf -> 0
    const float4* src = (const float4*)(bank + (size_t)idx * DIM);
    float4*       dst = (float4*)(out + (size_t)b * DIM);
    dst[threadIdx.x] = src[threadIdx.x];
}

// ---------------------------------------------------------------------------
extern "C" void kernel_launch(void* const* d_in, const int* in_sizes, int n_in,
                              void* d_out, int out_size, void* d_ws, size_t ws_size,
                              hipStream_t stream) {
    const float* feature = (const float*)d_in[0];
    const float* bank    = (const float*)d_in[1];
    const int*   clb     = (const int*)d_in[2];
    const int*   csb     = (const int*)d_in[3];
    const int*   cidx    = (const int*)d_in[4];
    const int*   gtl     = (const int*)d_in[5];
    float*       out     = (float*)d_out;

    unsigned long long* ws  = (unsigned long long*)d_ws;  // 512 u64 = 4KB
    unsigned long long* wsP = ws;
    unsigned long long* wsF = ws + B_ROWS;

    SCAN_init_kernel<<<1, 2 * B_ROWS, 0, stream>>>(ws);
    SCAN_main_kernel<<<NBLK, 512, 0, stream>>>(feature, bank, clb, csb, cidx, gtl,
                                               wsP, wsF);
    SCAN_gather_kernel<<<B_ROWS, DIM / 4, 0, stream>>>(bank, wsP, wsF, out);
}
